// Contrastive_loss_24283745091878
// MI455X (gfx1250) — compile-verified, matrix operand order checked by reading, not run on hardware
//
#include <hip/hip_runtime.h>
#include <hip/hip_bf16.h>

typedef __attribute__((ext_vector_type(16))) __bf16 v16bf;
typedef __attribute__((ext_vector_type(8)))  float  v8f;

// ---------------------------------------------------------------------------
// Problem constants
// ---------------------------------------------------------------------------
#define EMB 512
#define N2  512
#define KA  1280
#define KV  69120
#define VSAMP 138240            // 3*5*96*96 floats per visual sample
#define KSLICES 8
#define KV_SLICE (KV / KSLICES) // 8640, multiple of 32

#define LDS_STRIDE 40           // ushorts per 32-elem frag row (80B, padded)

__device__ __forceinline__ float wave_reduce_sum(float x) {
#pragma unroll
  for (int off = 16; off > 0; off >>= 1)
    x += __shfl_xor(x, off, 32);
  return x;
}

// ---- bf16 split helpers (bit-exact, storage-only __bf16) -------------------
__device__ __forceinline__ unsigned short bf16_rn_bits(float x) {
  unsigned u = __float_as_uint(x);
  return (unsigned short)((u + 0x7FFFu + ((u >> 16) & 1u)) >> 16);
}
__device__ __forceinline__ float bf16_bits_to_f32(unsigned short s) {
  return __uint_as_float(((unsigned)s) << 16);
}

__device__ __forceinline__ v16bf load_frag(const unsigned short* p) {
  union { uint4 q[2]; v16bf f; } u;
  u.q[0] = *(const uint4*)(p);
  u.q[1] = *(const uint4*)(p + 8);
  return u.f;
}
__device__ __forceinline__ void store_oct(unsigned short* p,
                                          const unsigned short* h) {
  uint4 q;
  q.x = (unsigned)h[0] | ((unsigned)h[1] << 16);
  q.y = (unsigned)h[2] | ((unsigned)h[3] << 16);
  q.z = (unsigned)h[4] | ((unsigned)h[5] << 16);
  q.w = (unsigned)h[6] | ((unsigned)h[7] << 16);
  *(uint4*)p = q;
}

// fragment-order offset for a K-octet base in {0,8,16,24}:
// frag element order per lane-half is K0..7,K16..23 | K8..15,K24..31
__device__ __forceinline__ int frag_oct_off(int koct) {
  return ((koct & 8) << 1) | ((koct & 16) >> 1) | (koct & ~24);
}

// ---------------------------------------------------------------------------
// bf16x3 split-precision WMMA GEMM:
//   C[512x512] (per K-slice) = A[512xK] * W[Kx512]
//   A,W split into hi/lo bf16 planes; acc += Ah*Bh + Ah*Bl + Al*Bh (f32 acc).
//   Macro tile 64x64 per 256-thread block; wave owns a 16x32 strip.
//   mode 0: contiguous audio rows; mode 1: visual gather (32-wide K block
//   always lies inside one contiguous 96-float run of the permuted tensor).
// ---------------------------------------------------------------------------
__global__ __launch_bounds__(256)
void gemm_wmma_bf16x3(const float* __restrict__ A0, const float* __restrict__ A1,
                      const float* __restrict__ W, float* __restrict__ Cout,
                      int K, int kcount, int mode)
{
  __shared__ __align__(16) unsigned short sAh[64 * LDS_STRIDE];
  __shared__ __align__(16) unsigned short sAl[64 * LDS_STRIDE];
  __shared__ __align__(16) unsigned short sBh[64 * LDS_STRIDE];
  __shared__ __align__(16) unsigned short sBl[64 * LDS_STRIDE];

  const int tid = threadIdx.x;
  const int bm = blockIdx.y, bn = blockIdx.x, bz = blockIdx.z;
  const int kb0 = bz * kcount;
  float* C = Cout + (size_t)bz * (EMB * N2);

  // ---- A loader: 64 rows x 32 k, 8 consecutive k per thread ----
  const int am  = tid >> 2;                 // 0..63
  const int akq = (tid & 3) * 8;            // 0,8,16,24
  const int afo = frag_oct_off(akq);        // frag-order element offset
  const int grow = bm * 64 + am;
  const float* arow_base;
  if (mode == 0)
    arow_base = (grow < 256) ? (A0 + (size_t)grow * K)
                             : (A1 + (size_t)(grow - 256) * K);
  else
    arow_base = (grow < 256) ? (A0 + (size_t)grow * VSAMP)
                             : (A1 + (size_t)(grow - 256) * VSAMP);

  // ---- B loader: K-octet per thread, N per lane (coalesced in N) ----
  const int bnn  = tid & 63;                // 0..63 (tile column)
  const int bkq  = (tid >> 6) * 8;          // 0,8,16,24
  const int bfo  = frag_oct_off(bkq);
  const float* wcol = W + (size_t)bn * 64 + bnn;

  // ---- compute mapping ----
  const int wv = tid >> 5, lane = tid & 31;
  const int lm = lane & 15, lh = lane >> 4;
  const int rt = wv >> 1, ct = wv & 1;      // wave tile rows rt*16, cols ct*32

  v8f acc0 = {}; v8f acc1 = {};

  for (int kb = kb0; kb < kb0 + kcount; kb += 32) {
    // ---------------- global loads ----------------
    const int k0 = kb + akq;
    const float* ap;
    if (mode == 0) {
      ap = arow_base + k0;
    } else {
      // flat k = ((t*3+c)*48+hh)*96+w  ->  src ((c*5+t)*96 + 48+hh)*96 + w
      unsigned tt = (unsigned)k0 / 13824u;            // 3*48*96
      unsigned r  = (unsigned)k0 - tt * 13824u;
      unsigned c  = r / 4608u;                        // 48*96
      unsigned r2 = r - c * 4608u;
      unsigned hh = r2 / 96u;
      unsigned w  = r2 - hh * 96u;
      ap = arow_base + (c * 5u + tt) * 9216u + (48u + hh) * 96u + w;
    }
    float4 av0 = *(const float4*)(ap);
    float4 av1 = *(const float4*)(ap + 4);

    float bx[8];
#pragma unroll
    for (int j = 0; j < 8; ++j)
      bx[j] = wcol[(size_t)(kb + bkq + j) * EMB];

    // ---------------- split + LDS stage ----------------
    float axs[8] = {av0.x, av0.y, av0.z, av0.w, av1.x, av1.y, av1.z, av1.w};
    unsigned short ah[8], al[8], bh[8], bl[8];
#pragma unroll
    for (int j = 0; j < 8; ++j) {
      ah[j] = bf16_rn_bits(axs[j]);
      al[j] = bf16_rn_bits(axs[j] - bf16_bits_to_f32(ah[j]));
      bh[j] = bf16_rn_bits(bx[j]);
      bl[j] = bf16_rn_bits(bx[j] - bf16_bits_to_f32(bh[j]));
    }

    __syncthreads();   // previous compute done before LDS overwrite
    store_oct(&sAh[am * LDS_STRIDE + afo], ah);
    store_oct(&sAl[am * LDS_STRIDE + afo], al);
    store_oct(&sBh[bnn * LDS_STRIDE + bfo], bh);
    store_oct(&sBl[bnn * LDS_STRIDE + bfo], bl);
    __syncthreads();

    // ---------------- fragments + WMMA ----------------
    const int arow  = (rt * 16 + lm) * LDS_STRIDE + lh * 16;
    const int brow0 = (ct * 32 + lm) * LDS_STRIDE + lh * 16;
    const int brow1 = brow0 + 16 * LDS_STRIDE;

    v16bf fah = load_frag(&sAh[arow]);
    v16bf fal = load_frag(&sAl[arow]);
    v16bf fbh0 = load_frag(&sBh[brow0]);
    v16bf fbl0 = load_frag(&sBl[brow0]);
    v16bf fbh1 = load_frag(&sBh[brow1]);
    v16bf fbl1 = load_frag(&sBl[brow1]);

    acc0 = __builtin_amdgcn_wmma_f32_16x16x32_bf16(false, fah, false, fbh0,
                                                   (short)0, acc0, false, false);
    acc1 = __builtin_amdgcn_wmma_f32_16x16x32_bf16(false, fah, false, fbh1,
                                                   (short)0, acc1, false, false);
    acc0 = __builtin_amdgcn_wmma_f32_16x16x32_bf16(false, fah, false, fbl0,
                                                   (short)0, acc0, false, false);
    acc1 = __builtin_amdgcn_wmma_f32_16x16x32_bf16(false, fah, false, fbl1,
                                                   (short)0, acc1, false, false);
    acc0 = __builtin_amdgcn_wmma_f32_16x16x32_bf16(false, fal, false, fbh0,
                                                   (short)0, acc0, false, false);
    acc1 = __builtin_amdgcn_wmma_f32_16x16x32_bf16(false, fal, false, fbh1,
                                                   (short)0, acc1, false, false);
  }

  // D layout: VGPR r -> M = r + 8*lh, N = lm
  const int colbase = bn * 64 + ct * 32 + lm;
  const int rowbase = bm * 64 + rt * 16 + lh * 8;
#pragma unroll
  for (int r = 0; r < 8; ++r) {
    C[(size_t)(rowbase + r) * EMB + colbase]      = acc0[r];
    C[(size_t)(rowbase + r) * EMB + colbase + 16] = acc1[r];
  }
}

// ---------------------------------------------------------------------------
// Row L2-normalization. Blocks 0..511: audio rows (ARAW). Blocks 512..1023:
// visual rows — sum the 8 split-K partials first.
// ---------------------------------------------------------------------------
__global__ __launch_bounds__(256)
void normalize_rows(const float* __restrict__ ARAW, const float* __restrict__ VP,
                    float* __restrict__ A, float* __restrict__ V)
{
  const int tid = threadIdx.x;
  const int rid = blockIdx.x;
  float e0, e1; float* dst;
  if (rid < N2) {
    e0 = ARAW[(size_t)rid * EMB + tid];
    e1 = ARAW[(size_t)rid * EMB + 256 + tid];
    dst = A + (size_t)rid * EMB;
  } else {
    const int row = rid - N2;
    e0 = 0.f; e1 = 0.f;
#pragma unroll
    for (int s = 0; s < KSLICES; ++s) {
      e0 += VP[(size_t)s * (EMB * N2) + (size_t)row * EMB + tid];
      e1 += VP[(size_t)s * (EMB * N2) + (size_t)row * EMB + 256 + tid];
    }
    dst = V + (size_t)row * EMB;
  }
  float ss = wave_reduce_sum(e0 * e0 + e1 * e1);
  __shared__ float red[8];
  if ((tid & 31) == 0) red[tid >> 5] = ss;
  __syncthreads();
  float tot = red[0] + red[1] + red[2] + red[3] +
              red[4] + red[5] + red[6] + red[7];
  float inv = 1.0f / fmaxf(sqrtf(tot), 1e-12f);
  dst[tid]       = e0 * inv;
  dst[tid + 256] = e1 * inv;
}

// ---------------------------------------------------------------------------
// Per-sample scoring: 6 numerator dots + 512-way denominator per sample.
// ---------------------------------------------------------------------------
__global__ __launch_bounds__(256)
void score_kernel(const float* __restrict__ A, const float* __restrict__ V,
                  float* __restrict__ lossi)
{
  const int i = blockIdx.x;
  const int tid = threadIdx.x;
  const int wv = tid >> 5, lane = tid & 31;
  __shared__ float sa1[EMB], sa2[EMB];
  __shared__ float sdot[6];
  __shared__ float sden[8];

  sa1[tid]       = A[(size_t)i * EMB + tid];
  sa1[tid + 256] = A[(size_t)i * EMB + tid + 256];
  sa2[tid]       = A[(size_t)(i + 256) * EMB + tid];
  sa2[tid + 256] = A[(size_t)(i + 256) * EMB + tid + 256];
  __syncthreads();

  if (wv < 6) {
    const float *x, *y;
    switch (wv) {
      case 0: x = sa1; y = V + (size_t)i * EMB;         break;
      case 1: x = sa1; y = V + (size_t)(i + 256) * EMB; break;
      case 2: x = sa2; y = V + (size_t)i * EMB;         break;
      case 3: x = sa2; y = V + (size_t)(i + 256) * EMB; break;
      case 4: x = sa1; y = sa2;                         break;
      default: x = V + (size_t)i * EMB; y = V + (size_t)(i + 256) * EMB; break;
    }
    float d = 0.f;
    for (int k = lane; k < EMB; k += 32) d += x[k] * y[k];
    d = wave_reduce_sum(d);
    if (lane == 0) sdot[wv] = d;
  }

  float den = 0.f;
  for (int j = wv; j < N2; j += 8) {
    const float* vj = V + (size_t)j * EMB;
    float d1 = 0.f, d2 = 0.f;
    for (int k = lane; k < EMB; k += 32) {
      float vv = vj[k];
      d1 += sa1[k] * vv;
      d2 += sa2[k] * vv;
    }
    d1 = wave_reduce_sum(d1);
    d2 = wave_reduce_sum(d2);
    den += __expf(d1) + __expf(d2);
  }
  if (lane == 0) sden[wv] = den;
  __syncthreads();

  if (tid == 0) {
    float dtot = sden[0] + sden[1] + sden[2] + sden[3] +
                 sden[4] + sden[5] + sden[6] + sden[7];
    float num = __expf(sdot[0]) + __expf(sdot[1]) + __expf(sdot[2]) +
                __expf(sdot[3]) + __expf(sdot[4]) + __expf(sdot[5]);
    lossi[i] = __logf(num) - __logf(dtot);
  }
}

__global__ __launch_bounds__(256)
void final_reduce(const float* __restrict__ lossi, float* __restrict__ out)
{
  const int tid = threadIdx.x;
  float v = wave_reduce_sum(lossi[tid]);
  __shared__ float red[8];
  if ((tid & 31) == 0) red[tid >> 5] = v;
  __syncthreads();
  if (tid == 0) {
    float s = red[0] + red[1] + red[2] + red[3] +
              red[4] + red[5] + red[6] + red[7];
    out[0] = -s / 256.0f;
  }
}

// ---------------------------------------------------------------------------
// Workspace: VP[8][512][512] | ARAW[512][512] | A | V | LI[256]  (~11.5 MB)
// ---------------------------------------------------------------------------
extern "C" void kernel_launch(void* const* d_in, const int* in_sizes, int n_in,
                              void* d_out, int out_size, void* d_ws, size_t ws_size,
                              hipStream_t stream)
{
  const float* a1 = (const float*)d_in[0];
  const float* v1 = (const float*)d_in[1];
  const float* a2 = (const float*)d_in[2];
  const float* v2 = (const float*)d_in[3];
  const float* Wa = (const float*)d_in[4];
  const float* Wv = (const float*)d_in[5];

  float* ws   = (float*)d_ws;
  float* VP   = ws;
  float* ARAW = VP + (size_t)KSLICES * EMB * N2;
  float* Am   = ARAW + (size_t)EMB * N2;
  float* Vm   = Am + (size_t)EMB * N2;
  float* LI   = Vm + (size_t)EMB * N2;

  dim3 blk(256);
  gemm_wmma_bf16x3<<<dim3(8, 8, 1), blk, 0, stream>>>(a1, a2, Wa, ARAW, KA, KA, 0);
  gemm_wmma_bf16x3<<<dim3(8, 8, KSLICES), blk, 0, stream>>>(v1, v2, Wv, VP, KV, KV_SLICE, 1);
  normalize_rows<<<dim3(2 * N2), blk, 0, stream>>>(ARAW, VP, Am, Vm);
  score_kernel<<<dim3(256), blk, 0, stream>>>(Am, Vm, LI);
  final_reduce<<<dim3(1), blk, 0, stream>>>(LI, (float*)d_out);
}
